// GraphMETNetwork_21500606283856
// MI455X (gfx1250) — compile-verified, matrix operand order checked by reading, not run on hardware
//
#include <hip/hip_runtime.h>
#include <hip/hip_bf16.h>
#include <math.h>

typedef __attribute__((ext_vector_type(16))) __bf16 v16bf;
typedef __attribute__((ext_vector_type(8)))  float  v8f;

#define HIDN 32
#define BN_EPS 1e-3f

__device__ __forceinline__ float eluf(float x) { return x > 0.f ? x : expm1f(x); }

// Native CDNA5 float atomic max (GLOBAL_ATOMIC_MAX_NUM_F32, non-returning).
__device__ __forceinline__ void atomicMaxF32(float* addr, float val) {
    asm volatile("global_atomic_max_num_f32 %0, %1, off scope:SCOPE_DEV"
                 :: "v"(addr), "v"(val)
                 : "memory");
}

// SADDR (GVS) form: 64-bit base lives in an SGPR pair; per-lane address is a
// 32-bit byte offset. Second column (+16 floats = 64 B) folded into the
// instruction's immediate offset field.
__device__ __forceinline__ void atomicMaxF32x2_s(const float* base, unsigned off32,
                                                 float v0, float v1) {
    asm volatile("global_atomic_max_num_f32 %0, %1, %3 scope:SCOPE_DEV\n\t"
                 "global_atomic_max_num_f32 %0, %2, %3 offset:64 scope:SCOPE_DEV"
                 :: "v"(off32), "v"(v0), "v"(v1), "s"(base)
                 : "memory");
}

// ---------------------------------------------------------------------------
// Stage 1: per-node feature encoder (cont MLP + embeddings + cat MLP + enc MLP)
// also accumulates per-column sum / sumsq for the first batch-norm.
// ---------------------------------------------------------------------------
__global__ __launch_bounds__(256) void k_embed(
    const float* __restrict__ x_cont, const int* __restrict__ x_cat,
    const float* __restrict__ datanorm,
    const float* __restrict__ W_cont, const float* __restrict__ b_cont,
    const float* __restrict__ emb_charge, const float* __restrict__ emb_pdg,
    const float* __restrict__ W_cat, const float* __restrict__ b_cat,
    const float* __restrict__ W_enc, const float* __restrict__ b_enc,
    float* __restrict__ embRaw, float* __restrict__ sum1, float* __restrict__ sum2,
    int N_)
{
    __shared__ float sS[HIDN], sQ[HIDN];
    if (threadIdx.x < HIDN) { sS[threadIdx.x] = 0.f; sQ[threadIdx.x] = 0.f; }
    __syncthreads();

    int n = blockIdx.x * blockDim.x + threadIdx.x;
    if (n < N_) {
        float xc[6];
        #pragma unroll
        for (int j = 0; j < 6; ++j) xc[j] = x_cont[n * 6 + j] * datanorm[j];

        float ec16[16];                       // emb_cont
        #pragma unroll
        for (int h = 0; h < 16; ++h) {
            float a = b_cont[h];
            #pragma unroll
            for (int j = 0; j < 6; ++j) a += xc[j] * W_cont[j * 16 + h];
            ec16[h] = eluf(a);
        }

        int pdg = abs(x_cat[2 * n]);
        int chg = x_cat[2 * n + 1];
        int pi = (pdg == 1) ? 0 : (pdg == 2) ? 1 : (pdg == 11) ? 2 : (pdg == 13) ? 3
               : (pdg == 22) ? 4 : (pdg == 130) ? 5 : 6;

        float cat16[16];
        #pragma unroll
        for (int k = 0; k < 8; ++k) cat16[k]     = emb_charge[(chg + 1) * 8 + k];
        #pragma unroll
        for (int k = 0; k < 8; ++k) cat16[8 + k] = emb_pdg[pi * 8 + k];

        float in32[32];
        #pragma unroll
        for (int h = 0; h < 16; ++h) {        // emb_cat MLP -> first 16 lanes
            float a = b_cat[h];
            #pragma unroll
            for (int k = 0; k < 16; ++k) a += cat16[k] * W_cat[k * 16 + h];
            in32[h] = eluf(a);
        }
        #pragma unroll
        for (int h = 0; h < 16; ++h) in32[16 + h] = ec16[h];

        #pragma unroll
        for (int h = 0; h < HIDN; ++h) {
            float a = b_enc[h];
            #pragma unroll
            for (int k = 0; k < HIDN; ++k) a += in32[k] * W_enc[k * HIDN + h];
            float e = eluf(a);
            embRaw[n * HIDN + h] = e;
            atomicAdd(&sS[h], e);
            atomicAdd(&sQ[h], e * e);
        }
    }
    __syncthreads();
    if (threadIdx.x < HIDN) {
        atomicAdd(&sum1[threadIdx.x], sS[threadIdx.x]);
        atomicAdd(&sum2[threadIdx.x], sQ[threadIdx.x]);
    }
}

// ---------------------------------------------------------------------------
// Stage 2: apply batch-norm; write f32 copy (residual path) + bf16 copy
// (gather/WMMA path — halves the L2 gather traffic in the edge kernel).
// ---------------------------------------------------------------------------
__global__ __launch_bounds__(256) void k_bn_apply(
    const float* __restrict__ embRaw,
    const float* __restrict__ sum1, const float* __restrict__ sum2,
    const float* __restrict__ g, const float* __restrict__ be,
    float* __restrict__ embN, __bf16* __restrict__ embBf, int N_)
{
    int idx = blockIdx.x * blockDim.x + threadIdx.x;
    if (idx >= N_ * HIDN) return;
    int col = idx & (HIDN - 1);
    float invN = 1.f / (float)N_;
    float m = sum1[col] * invN;
    float v = sum2[col] * invN - m * m;
    float y = g[col] * (embRaw[idx] - m) * rsqrtf(v + BN_EPS) + be[col];
    embN[idx]  = y;
    embBf[idx] = (__bf16)y;
}

__global__ __launch_bounds__(256) void k_fill(float* __restrict__ p, float v, int cnt) {
    int idx = blockIdx.x * blockDim.x + threadIdx.x;
    if (idx < cnt) p[idx] = v;
}

__global__ __launch_bounds__(256) void k_tobf(const float* __restrict__ s,
                                              __bf16* __restrict__ d, int cnt) {
    int idx = blockIdx.x * blockDim.x + threadIdx.x;
    if (idx < cnt) d[idx] = (__bf16)s[idx];
}

// ---------------------------------------------------------------------------
// Stage 3: edge kernel. One wave = 16 edges. A (16x64 bf16) gathered per tile,
// B (64x32 bf16 = W_msg) held in registers for the whole grid-stride loop.
// D = A @ B + b_msg via 4x v_wmma_f32_16x16x32_bf16, then native
// global_atomic_max_num_f32 (SADDR form, 32-bit offsets) segment-max scatter.
//
// The main loop covers FULL tiles only (no guards, no divergence). The <=15
// leftover edges are handled by a scalar VALU path on wave 0.
//
// 16-bit A layout (ISA 7.12.2): lanes 0-15 -> M=lane, K pairs
//   vgpr v<4: K=2v,2v+1 ; v>=4: K=2v+8,2v+9 ; lanes 16-31 add +8 to K.
// B mirrored with N across lanes 0-15.
// ---------------------------------------------------------------------------
__global__ __launch_bounds__(256) void k_edges(
    const int* __restrict__ srcIdx, const int* __restrict__ dstIdx,
    const __bf16* __restrict__ embBf,
    const __bf16* __restrict__ WmsgBf, const float* __restrict__ b_msg,
    float* __restrict__ agg, int E_, int nFull)
{
    const int lane  = threadIdx.x & 31;
    const int mlo   = lane & 15;
    const int hi    = lane >> 4;
    const int gwave = blockIdx.x * (blockDim.x >> 5) + (threadIdx.x >> 5);
    const int nWav  = gridDim.x * (blockDim.x >> 5);

    // Preload the four 32x16 blocks of W_msg (K-chunk x N-chunk) into registers.
    v16bf B00, B01, B10, B11;
    {
        int n0 = mlo, n1 = 16 + mlo;
        #pragma unroll
        for (int v = 0; v < 8; ++v) {
            int kl = ((v < 4) ? 2 * v : 2 * v + 8) + 8 * hi;
            B00[2 * v]     = WmsgBf[(kl)      * HIDN + n0];
            B00[2 * v + 1] = WmsgBf[(kl + 1)  * HIDN + n0];
            B01[2 * v]     = WmsgBf[(kl)      * HIDN + n1];
            B01[2 * v + 1] = WmsgBf[(kl + 1)  * HIDN + n1];
            B10[2 * v]     = WmsgBf[(32 + kl)     * HIDN + n0];
            B10[2 * v + 1] = WmsgBf[(32 + kl + 1) * HIDN + n0];
            B11[2 * v]     = WmsgBf[(32 + kl)     * HIDN + n1];
            B11[2 * v + 1] = WmsgBf[(32 + kl + 1) * HIDN + n1];
        }
    }
    const float    bias0 = b_msg[mlo];
    const float    bias1 = b_msg[16 + mlo];
    const unsigned colB  = (unsigned)(mlo * 4);     // byte offset of my column

    for (int t = gwave; t < nFull; t += nWav) {
        const int ebase = t * 16;
        const int sI = srcIdx[ebase + mlo];
        const int dI = dstIdx[ebase + mlo];

        // dst ids for my 8 scatter rows (m = r + 8*hi): one cacheline, two
        // int4 loads issued early so latency hides under gather + WMMA.
        const int ib = ebase + 8 * hi;
        const int4 dA = *(const int4*)(dstIdx + ib);
        const int4 dB = *(const int4*)(dstIdx + ib + 4);
        const int dd[8] = { dA.x, dA.y, dA.z, dA.w, dB.x, dB.y, dB.z, dB.w };

        // Speculative prefetch of next tile's indices (global_prefetch_b8).
        __builtin_prefetch(dstIdx + (size_t)(t + nWav) * 16 + mlo, 0, 1);

        const __bf16* xi = embBf + (size_t)dI * HIDN;   // x_i = emb[dst]
        const __bf16* xj = embBf + (size_t)sI * HIDN;   // x_j = emb[src]

        v16bf A0, A1;                                    // K 0..31 : x_i ; K 32..63 : x_j - x_i
        #pragma unroll
        for (int v = 0; v < 8; ++v) {
            int kl = ((v < 4) ? 2 * v : 2 * v + 8) + 8 * hi;
            __bf16 a = xi[kl], b = xi[kl + 1];
            A0[2 * v]     = a;
            A0[2 * v + 1] = b;
            A1[2 * v]     = (__bf16)((float)xj[kl]     - (float)a);
            A1[2 * v + 1] = (__bf16)((float)xj[kl + 1] - (float)b);
        }

        v8f c0, c1;
        #pragma unroll
        for (int r = 0; r < 8; ++r) { c0[r] = bias0; c1[r] = bias1; }

        c0 = __builtin_amdgcn_wmma_f32_16x16x32_bf16(false, A0, false, B00, (short)0, c0, false, false);
        c0 = __builtin_amdgcn_wmma_f32_16x16x32_bf16(false, A1, false, B10, (short)0, c0, false, false);
        c1 = __builtin_amdgcn_wmma_f32_16x16x32_bf16(false, A0, false, B01, (short)0, c1, false, false);
        c1 = __builtin_amdgcn_wmma_f32_16x16x32_bf16(false, A1, false, B11, (short)0, c1, false, false);

        // C layout: lane column = mlo (c0) / mlo+16 (c1); vgpr r -> row m = r + 8*hi.
        // 32-bit byte offsets against the SGPR-resident agg base.
        #pragma unroll
        for (int r = 0; r < 8; ++r)
            atomicMaxF32x2_s(agg, (unsigned)dd[r] * (HIDN * 4u) + colB, c0[r], c1[r]);
    }

    // Scalar tail: edges [nFull*16, E_), one wave, one output column per lane.
    if (gwave == 0) {
        for (int e = nFull * 16; e < E_; ++e) {
            const int s = srcIdx[e], d = dstIdx[e];
            float acc = b_msg[lane];
            #pragma unroll
            for (int k = 0; k < HIDN; ++k) {
                float xik = (float)embBf[(size_t)d * HIDN + k];
                float xjk = (float)embBf[(size_t)s * HIDN + k];
                acc += xik * (float)WmsgBf[k * HIDN + lane]
                     + (xjk - xik) * (float)WmsgBf[(HIDN + k) * HIDN + lane];
            }
            atomicMaxF32(&agg[(size_t)d * HIDN + lane], acc);
        }
    }
}

// ---------------------------------------------------------------------------
// Stage 4: replace -inf (isolated nodes) with 0 in-place + BN sums for conv BN.
// ---------------------------------------------------------------------------
__global__ __launch_bounds__(256) void k_agg_stats(
    float* __restrict__ agg, float* __restrict__ sum1, float* __restrict__ sum2, int N_)
{
    __shared__ float sS[HIDN], sQ[HIDN];
    if (threadIdx.x < HIDN) { sS[threadIdx.x] = 0.f; sQ[threadIdx.x] = 0.f; }
    __syncthreads();
    int idx = blockIdx.x * blockDim.x + threadIdx.x;
    if (idx < N_ * HIDN) {
        float v = agg[idx];
        if (isinf(v) && v < 0.f) { v = 0.f; agg[idx] = 0.f; }
        int col = idx & (HIDN - 1);
        atomicAdd(&sS[col], v);
        atomicAdd(&sQ[col], v * v);
    }
    __syncthreads();
    if (threadIdx.x < HIDN) {
        atomicAdd(&sum1[threadIdx.x], sS[threadIdx.x]);
        atomicAdd(&sum2[threadIdx.x], sQ[threadIdx.x]);
    }
}

// ---------------------------------------------------------------------------
// Stage 5: residual add of BN'd aggregation + output MLP (32->16 ELU ->1).
// ---------------------------------------------------------------------------
__global__ __launch_bounds__(256) void k_final(
    const float* __restrict__ embN, const float* __restrict__ agg,
    const float* __restrict__ sum1, const float* __restrict__ sum2,
    const float* __restrict__ g_conv, const float* __restrict__ be_conv,
    const float* __restrict__ W_o1, const float* __restrict__ b_o1,
    const float* __restrict__ W_o2, const float* __restrict__ b_o2,
    float* __restrict__ out, int N_)
{
    int n = blockIdx.x * blockDim.x + threadIdx.x;
    if (n >= N_) return;
    float invN = 1.f / (float)N_;
    float h[HIDN];
    #pragma unroll
    for (int k = 0; k < HIDN; ++k) {
        float m = sum1[k] * invN;
        float v = sum2[k] * invN - m * m;
        float a = g_conv[k] * (agg[n * HIDN + k] - m) * rsqrtf(v + BN_EPS) + be_conv[k];
        h[k] = embN[n * HIDN + k] + a;
    }
    float acc = b_o2[0];
    #pragma unroll
    for (int j = 0; j < 16; ++j) {
        float o = b_o1[j];
        #pragma unroll
        for (int k = 0; k < HIDN; ++k) o += h[k] * W_o1[k * 16 + j];
        acc += eluf(o) * W_o2[j];
    }
    out[n] = acc;
}

// ---------------------------------------------------------------------------
extern "C" void kernel_launch(void* const* d_in, const int* in_sizes, int n_in,
                              void* d_out, int out_size, void* d_ws, size_t ws_size,
                              hipStream_t stream)
{
    const float* x_cont     = (const float*)d_in[0];
    const int*   x_cat      = (const int*)  d_in[1];
    const int*   edge_index = (const int*)  d_in[2];
    const float* datanorm   = (const float*)d_in[4];
    const float* W_cont     = (const float*)d_in[5];
    const float* b_cont     = (const float*)d_in[6];
    const float* emb_charge = (const float*)d_in[7];
    const float* emb_pdg    = (const float*)d_in[8];
    const float* W_cat      = (const float*)d_in[9];
    const float* b_cat      = (const float*)d_in[10];
    const float* W_enc      = (const float*)d_in[11];
    const float* b_enc      = (const float*)d_in[12];
    const float* g_all      = (const float*)d_in[13];
    const float* be_all     = (const float*)d_in[14];
    const float* W_msg      = (const float*)d_in[15];
    const float* b_msg      = (const float*)d_in[16];
    const float* g_conv     = (const float*)d_in[17];
    const float* be_conv    = (const float*)d_in[18];
    const float* W_o1       = (const float*)d_in[19];
    const float* b_o1       = (const float*)d_in[20];
    const float* W_o2       = (const float*)d_in[21];
    const float* b_o2       = (const float*)d_in[22];

    const int N_ = in_sizes[3];          // batch has N elements
    const int E_ = in_sizes[2] / 2;      // edge_index is 2 x E
    const int* srcIdx = edge_index;      // edge_index[0]
    const int* dstIdx = edge_index + E_; // edge_index[1]

    // Workspace carving (256B aligned slices)
    char*  base = (char*)d_ws;
    size_t off  = 0;
    auto carve = [&](size_t bytes) -> void* {
        void* p = base + off;
        off += (bytes + 255) & ~(size_t)255;
        return p;
    };
    float*  embRaw = (float*)carve((size_t)N_ * HIDN * sizeof(float));
    float*  embN   = (float*)carve((size_t)N_ * HIDN * sizeof(float));
    __bf16* embBf  = (__bf16*)carve((size_t)N_ * HIDN * sizeof(__bf16));
    float*  agg    = (float*)carve((size_t)N_ * HIDN * sizeof(float));
    __bf16* WmsgBf = (__bf16*)carve((size_t)64 * HIDN * sizeof(__bf16));
    float*  stats  = (float*)carve(4 * HIDN * sizeof(float));
    float*  eS1 = stats, *eS2 = stats + HIDN, *aS1 = stats + 2 * HIDN, *aS2 = stats + 3 * HIDN;

    (void)hipMemsetAsync(stats, 0, 4 * HIDN * sizeof(float), stream);

    const int TB = 256;
    const int nodeBlocks = (N_ + TB - 1) / TB;
    const int elemBlocks = (N_ * HIDN + TB - 1) / TB;

    k_embed<<<nodeBlocks, TB, 0, stream>>>(x_cont, x_cat, datanorm, W_cont, b_cont,
                                           emb_charge, emb_pdg, W_cat, b_cat,
                                           W_enc, b_enc, embRaw, eS1, eS2, N_);

    k_bn_apply<<<elemBlocks, TB, 0, stream>>>(embRaw, eS1, eS2, g_all, be_all,
                                              embN, embBf, N_);

    // -inf fill value computed host-side (bit pattern 0xFF800000)
    const float NEG_INF = -__builtin_inff();
    k_fill<<<elemBlocks, TB, 0, stream>>>(agg, NEG_INF, N_ * HIDN);
    k_tobf<<<(64 * HIDN + TB - 1) / TB, TB, 0, stream>>>(W_msg, WmsgBf, 64 * HIDN);

    const int nFull = E_ / 16;           // full 16-edge tiles
    int edgeBlocks = 1024;
    if (edgeBlocks * 8 > nFull) edgeBlocks = (nFull + 7) / 8;
    if (edgeBlocks < 1) edgeBlocks = 1;
    k_edges<<<edgeBlocks, TB, 0, stream>>>(srcIdx, dstIdx, embBf, WmsgBf, b_msg,
                                           agg, E_, nFull);

    k_agg_stats<<<elemBlocks, TB, 0, stream>>>(agg, aS1, aS2, N_);

    k_final<<<nodeBlocks, TB, 0, stream>>>(embN, agg, aS1, aS2, g_conv, be_conv,
                                           W_o1, b_o1, W_o2, b_o2, (float*)d_out, N_);
}
